// ContextEncoderModel_71184787964201
// MI455X (gfx1250) — matrix operand order repaired and checked by reading, not compiled, not run on hardware
//
#include <hip/hip_runtime.h>
#include <hip/hip_bf16.h>
#include <math.h>

// ---------------------------------------------------------------------------
// Problem constants (match reference)
// ---------------------------------------------------------------------------
constexpr int Bc = 256;   // batch
constexpr int Tc = 128;   // time steps
constexpr int Ec = 300;   // embed dim
constexpr int Hc = 512;   // LSTM hidden
constexpr int Dc = 1024;  // MLP dim
constexpr int XK = 320;   // padded embed dim (multiple of 32)
constexpr int KT = 26;    // K tiles per LSTM step GEMM: (320+512)/32
constexpr int XT = XK / 32;              // 10 k-tiles sourced from embeddings
constexpr int HT = KT - XT;              // 16 k-tiles sourced from h
constexpr int GSTRIDE = 32 * KT * 512;   // elems between gate blocks in packed W
constexpr int MKT = 32;   // K tiles for MLP GEMMs (1024/32)

typedef __attribute__((ext_vector_type(16))) __bf16 v16bf;
typedef __attribute__((ext_vector_type(8)))  __bf16 v8bf;
typedef __attribute__((ext_vector_type(8)))  float  v8f;

__device__ __forceinline__ float sigmf(float x) {
    return 1.0f / (1.0f + __expf(-x));
}

__device__ __forceinline__ v16bf load_a_frag(const __bf16* src, int half) {
    v8bf lo = *(const v8bf*)(src + 16 * half);        // global_load_b128
    v8bf hi = *(const v8bf*)(src + 16 * half + 8);   // global_load_b128
    return __builtin_shufflevector(lo, hi, 0, 1, 2, 3, 4, 5, 6, 7,
                                   8, 9, 10, 11, 12, 13, 14, 15);
}

// ---------------------------------------------------------------------------
// One-time data staging kernels
// ---------------------------------------------------------------------------

// Xbf[side][b][t][0..319] : bf16 embeddings, zero-padded 300->320.
__global__ void build_xbf_kernel(const float* __restrict__ L,
                                 const float* __restrict__ R,
                                 __bf16* __restrict__ Xbf) {
    size_t idx = (size_t)blockIdx.x * blockDim.x + threadIdx.x;
    size_t total = (size_t)2 * Bc * Tc * XK;
    if (idx >= total) return;
    int k = (int)(idx % XK);
    size_t bt = idx / XK;
    int t = (int)(bt % Tc);
    size_t sb = bt / Tc;
    int b = (int)(sb % Bc);
    int side = (int)(sb / Bc);
    const float* src = side ? R : L;
    float v = (k < Ec) ? src[((size_t)b * Tc + t) * Ec + k] : 0.0f;
    Xbf[idx] = (__bf16)v;
}

// Pack LSTM weights [812, 2048] f32 into WMMA B-matrix lane layout, bf16,
// with K padded 300->320 (zeros) before the recurrent rows.
// Wpack[side][nt(0..127)][kt(0..25)][lane(0..31)][e(0..15)]
//   element: k = kt*32 + 16*(lane>>4) + e ; n = nt*16 + (lane&15)
__global__ void pack_lstm_w_kernel(const float* __restrict__ Wl,
                                   const float* __restrict__ Wr,
                                   __bf16* __restrict__ out) {
    const int lane = threadIdx.x;          // 32
    const int kt = blockIdx.x % KT;
    const int nt = blockIdx.x / KT;        // 0..127
    const int side = blockIdx.z;
    const float* W = side ? Wr : Wl;
    const int n = nt * 16 + (lane & 15);
    const int kb = kt * 32 + 16 * (lane >> 4);
    v16bf v;
#pragma unroll
    for (int e = 0; e < 16; ++e) {
        int k = kb + e;
        float f = 0.0f;
        if (k < Ec)        f = W[(size_t)k * (4 * Hc) + n];
        else if (k >= XK)  f = W[(size_t)(k - (XK - Ec)) * (4 * Hc) + n];
        v[e] = (__bf16)f;
    }
    *(v16bf*)(out + ((((size_t)side * 128 + nt) * KT + kt) * 32 + lane) * 16) = v;
}

// Pack a [1024, 1024] f32 weight into WMMA B layout, bf16.
// Mpack[nt(0..63)][kt(0..31)][lane][e]
__global__ void pack_mlp_w_kernel(const float* __restrict__ W,
                                  __bf16* __restrict__ out) {
    const int lane = threadIdx.x;          // 32
    const int kt = blockIdx.x & 31;
    const int nt = blockIdx.x >> 5;        // 0..63
    const int n = nt * 16 + (lane & 15);
    const int kb = kt * 32 + 16 * (lane >> 4);
    v16bf v;
#pragma unroll
    for (int e = 0; e < 16; ++e)
        v[e] = (__bf16)W[(size_t)(kb + e) * Dc + n];
    *(v16bf*)(out + (((size_t)nt * MKT + kt) * 32 + lane) * 16) = v;
}

// Zero c state (f32) and h state (bf16) for both sides.
__global__ void init_state_kernel(float* __restrict__ Cst,
                                  __bf16* __restrict__ Hbf) {
    int idx = blockIdx.x * blockDim.x + threadIdx.x;
    if (idx < 2 * Bc * Hc) {
        Cst[idx] = 0.0f;
        Hbf[idx] = (__bf16)0.0f;
    }
}

// ---------------------------------------------------------------------------
// Fused LSTM step: gates GEMM (WMMA, bf16 in / f32 acc) + pointwise update.
// grid (8, 8, 2), block 128 (4 waves).
//   wave w handles ntile = blockIdx.x*4+w (0..31) -> h columns [ntile*16, +16)
//   and TWO row tiles (btiles 2*blockIdx.y, +1): W tiles loaded once feed 8
//   WMMAs (i/j/f/o x 2 row tiles), halving W traffic and hiding load latency.
// K loop is split: k-tiles 0..9 read embeddings, 10..25 read h(t-1); no
// per-iteration source select. Length mask freezes c/h past sequence end.
// ---------------------------------------------------------------------------
__global__ __launch_bounds__(128) void lstm_step_kernel(
    const __bf16* __restrict__ Xbf, const __bf16* __restrict__ Wpack,
    const float* __restrict__ bL, const float* __restrict__ bR,
    const int* __restrict__ lenL, const int* __restrict__ lenR,
    float* __restrict__ Cst, __bf16* __restrict__ Hbf, int t) {
    const int lane = threadIdx.x & 31;
    const int wave = threadIdx.x >> 5;
    const int ntile = blockIdx.x * 4 + wave;   // 0..31
    const int btile = blockIdx.y * 2;          // 0,2,..,14
    const int side = blockIdx.z;
    const float* bias = side ? bR : bL;
    const int* lens = side ? lenR : lenL;

    // A-fragment addressing (16-bit A 16x32 layout):
    //   lane<16 holds K 0-15 of the tile, lane>=16 holds K 16-31 (contiguous)
    const int m = lane & 15;
    const int half = lane >> 4;
    const int arow0 = btile * 16 + m;
    const int arow1 = arow0 + 16;
    const __bf16* xrow0 = Xbf + (((size_t)side * Bc + arow0) * Tc + t) * XK;
    const __bf16* xrow1 = Xbf + (((size_t)side * Bc + arow1) * Tc + t) * XK;
    const __bf16* hrow0 = Hbf + ((size_t)side * Bc + arow0) * Hc;
    const __bf16* hrow1 = Hbf + ((size_t)side * Bc + arow1) * Hc;

    // Packed W walk: 512 elems per k-tile, gate blocks at +g*GSTRIDE.
    const __bf16* wp = Wpack +
        (((size_t)side * 128 + ntile) * KT * 32 + lane) * 16;

    v8f acc[2][4];
#pragma unroll
    for (int s = 0; s < 2; ++s)
#pragma unroll
        for (int g = 0; g < 4; ++g) acc[s][g] = (v8f){};

    // ---- Phase 1: K from embeddings (k-tiles 0..9) ----
    for (int kt = 0; kt < XT; ++kt) {
        v16bf a0 = load_a_frag(xrow0 + kt * 32, half);
        v16bf a1 = load_a_frag(xrow1 + kt * 32, half);
        const __bf16* w = wp + (size_t)kt * 512;
#pragma unroll
        for (int g = 0; g < 4; ++g) {
            v16bf wg = *(const v16bf*)(w + (size_t)g * GSTRIDE);
            acc[0][g] = __builtin_amdgcn_wmma_f32_16x16x32_bf16(
                false, a0, false, wg, (short)0, acc[0][g], false, false);
            acc[1][g] = __builtin_amdgcn_wmma_f32_16x16x32_bf16(
                false, a1, false, wg, (short)0, acc[1][g], false, false);
        }
    }

    // ---- Phase 2: K from h(t-1) (k-tiles 10..25) ----
    for (int kk = 0; kk < HT; ++kk) {
        v16bf a0 = load_a_frag(hrow0 + kk * 32, half);
        v16bf a1 = load_a_frag(hrow1 + kk * 32, half);
        const __bf16* w = wp + (size_t)(XT + kk) * 512;
#pragma unroll
        for (int g = 0; g < 4; ++g) {
            v16bf wg = *(const v16bf*)(w + (size_t)g * GSTRIDE);
            acc[0][g] = __builtin_amdgcn_wmma_f32_16x16x32_bf16(
                false, a0, false, wg, (short)0, acc[0][g], false, false);
            acc[1][g] = __builtin_amdgcn_wmma_f32_16x16x32_bf16(
                false, a1, false, wg, (short)0, acc[1][g], false, false);
        }
    }

    // C/D tile layout: n = lane&15; rows rbase..rbase+7 where rbase = 8*(lane>>4)
    const int n = ntile * 16 + (lane & 15);
    const int rbase = 8 * (lane >> 4);
    const float bi = bias[n];
    const float bj = bias[n + Hc];
    const float bf_ = bias[n + 2 * Hc] + 1.0f;   // FORGET_BIAS
    const float bo = bias[n + 3 * Hc];

#pragma unroll
    for (int s = 0; s < 2; ++s) {
#pragma unroll
        for (int r = 0; r < 8; ++r) {
            const int brow = (btile + s) * 16 + rbase + r;
            const size_t idx = ((size_t)side * Bc + brow) * Hc + n;
            const float cold = Cst[idx];
            const float iv = sigmf(acc[s][0][r] + bi);
            const float jv = tanhf(acc[s][1][r] + bj);
            const float fv = sigmf(acc[s][2][r] + bf_);
            const float ov = sigmf(acc[s][3][r] + bo);
            const float nc = cold * fv + iv * jv;
            const float nh = tanhf(nc) * ov;
            if (t < lens[brow]) {       // freeze state past sequence length
                Cst[idx] = nc;
                Hbf[idx] = (__bf16)nh;
            }
        }
    }
}

// Concatenate final hidden states: Xc[b][0..511]=h_left, [512..1023]=h_right
__global__ void concat_kernel(const __bf16* __restrict__ Hbf,
                              __bf16* __restrict__ Xc) {
    int idx = blockIdx.x * blockDim.x + threadIdx.x;   // B * 1024
    if (idx >= Bc * Dc) return;
    int k = idx & (Dc - 1);
    int b = idx >> 10;
    int side = k >> 9;
    int kk = k & (Hc - 1);
    Xc[idx] = Hbf[((size_t)side * Bc + b) * Hc + kk];
}

// ---------------------------------------------------------------------------
// MLP layer: relu([256,1024] @ [1024,1024]) via WMMA. grid (16,16), block 128.
// outF != nullptr => final layer, write f32 to d_out; else bf16 intermediate.
// ---------------------------------------------------------------------------
__global__ __launch_bounds__(128) void mlp_gemm_kernel(
    const __bf16* __restrict__ A, const __bf16* __restrict__ Wp,
    __bf16* __restrict__ outBf, float* __restrict__ outF) {
    const int lane = threadIdx.x & 31;
    const int wave = threadIdx.x >> 5;
    const int ntile = blockIdx.x * 4 + wave;   // 0..63
    const int btile = blockIdx.y;              // 0..15
    const int m = lane & 15;
    const int half = lane >> 4;
    const __bf16* arp = A + (size_t)(btile * 16 + m) * Dc;

    v8f acc = {};
    for (int kt = 0; kt < MKT; ++kt) {
        v16bf a = load_a_frag(arp + kt * 32, half);
        v16bf w = *(const v16bf*)(Wp + (((size_t)ntile * MKT + kt) * 32 + lane) * 16);
        acc = __builtin_amdgcn_wmma_f32_16x16x32_bf16(false, a, false, w,
                                                      (short)0, acc, false, false);
    }

    const int n = ntile * 16 + (lane & 15);
    const int rbase = 8 * (lane >> 4);
#pragma unroll
    for (int r = 0; r < 8; ++r) {
        const int row = btile * 16 + rbase + r;
        float v = acc[r];
        v = v > 0.0f ? v : 0.0f;                 // ReLU
        if (outF) outF[(size_t)row * Dc + n] = v;
        else      outBf[(size_t)row * Dc + n] = (__bf16)v;
    }
}

// ---------------------------------------------------------------------------
// Host launcher
// ---------------------------------------------------------------------------
extern "C" void kernel_launch(void* const* d_in, const int* in_sizes, int n_in,
                              void* d_out, int out_size, void* d_ws, size_t ws_size,
                              hipStream_t stream) {
    (void)in_sizes; (void)n_in; (void)out_size; (void)ws_size;
    const float* left_embed  = (const float*)d_in[0];
    const float* right_embed = (const float*)d_in[1];
    const int*   left_len    = (const int*)d_in[2];
    const int*   right_len   = (const int*)d_in[3];
    const float* W_left      = (const float*)d_in[4];
    const float* b_left      = (const float*)d_in[5];
    const float* W_right     = (const float*)d_in[6];
    const float* b_right     = (const float*)d_in[7];
    const float* trans_w     = (const float*)d_in[8];
    const float* hidden_ws   = (const float*)d_in[9];
    float* out = (float*)d_out;

    char* ws = (char*)d_ws;
    size_t off = 0;
    auto take = [&](size_t bytes) -> char* {
        char* p = ws + off;
        off = (off + bytes + 255) & ~(size_t)255;
        return p;
    };
    __bf16* Xbf   = (__bf16*)take((size_t)2 * Bc * Tc * XK * 2);      // ~42 MB
    __bf16* Wpack = (__bf16*)take((size_t)2 * 128 * KT * 512 * 2);    // ~6.8 MB
    __bf16* Mp0   = (__bf16*)take((size_t)64 * MKT * 512 * 2);        // 2 MB
    __bf16* Mp1   = (__bf16*)take((size_t)64 * MKT * 512 * 2);
    __bf16* Mp2   = (__bf16*)take((size_t)64 * MKT * 512 * 2);
    float*  Cst   = (float*) take((size_t)2 * Bc * Hc * 4);           // 1 MB
    __bf16* Hbf   = (__bf16*)take((size_t)2 * Bc * Hc * 2);
    __bf16* Xc    = (__bf16*)take((size_t)Bc * Dc * 2);
    __bf16* Yb    = (__bf16*)take((size_t)Bc * Dc * 2);

    // Stage data (re-done every call: deterministic, no cross-call state).
    {
        size_t total = (size_t)2 * Bc * Tc * XK;
        build_xbf_kernel<<<(unsigned)((total + 255) / 256), 256, 0, stream>>>(
            left_embed, right_embed, Xbf);
    }
    pack_lstm_w_kernel<<<dim3(128 * KT, 1, 2), 32, 0, stream>>>(W_left, W_right, Wpack);
    pack_mlp_w_kernel<<<64 * MKT, 32, 0, stream>>>(trans_w, Mp0);
    pack_mlp_w_kernel<<<64 * MKT, 32, 0, stream>>>(hidden_ws, Mp1);
    pack_mlp_w_kernel<<<64 * MKT, 32, 0, stream>>>(hidden_ws + (size_t)Dc * Dc, Mp2);
    init_state_kernel<<<(2 * Bc * Hc + 255) / 256, 256, 0, stream>>>(Cst, Hbf);

    // Sequential recurrence: one fused GEMM+pointwise launch per step;
    // both LSTMs run concurrently via gridDim.z.
    for (int t = 0; t < Tc; ++t) {
        lstm_step_kernel<<<dim3(8, 8, 2), 128, 0, stream>>>(
            Xbf, Wpack, b_left, b_right, left_len, right_len, Cst, Hbf, t);
    }

    // MLP head.
    concat_kernel<<<(Bc * Dc + 255) / 256, 256, 0, stream>>>(Hbf, Xc);
    mlp_gemm_kernel<<<dim3(16, 16), 128, 0, stream>>>(Xc, Mp0, Yb, nullptr);
    mlp_gemm_kernel<<<dim3(16, 16), 128, 0, stream>>>(Yb, Mp1, Xc, nullptr);
    mlp_gemm_kernel<<<dim3(16, 16), 128, 0, stream>>>(Xc, Mp2, nullptr, out);
}
